// RNN_Attn_37976100831494
// MI455X (gfx1250) — matrix operand order repaired and checked by reading, not compiled, not run on hardware
//
#include <hip/hip_runtime.h>
#include <cstdint>
#include <cstddef>

// ---------------------------------------------------------------------------
// Types / helpers
// ---------------------------------------------------------------------------
typedef unsigned short u16;
typedef __attribute__((ext_vector_type(16))) __bf16 bf16x16;
typedef __attribute__((ext_vector_type(8)))  float  v8f;

union Frag { bf16x16 v; u16 u[16]; unsigned int d[8]; };

enum { kB = 64, kF = 2048, kP = 196, kT = 25, kE = 512, kH = 512, kA = 512, kV = 30000 };

__device__ __forceinline__ u16 f2bf(float f) {
    unsigned int u = __float_as_uint(f);
    unsigned int r = u + 0x7FFFu + ((u >> 16) & 1u);   // round-to-nearest-even
    return (u16)(r >> 16);
}
__device__ __forceinline__ float bf2f(u16 h) {
    return __uint_as_float(((unsigned int)h) << 16);
}

// ---------------------------------------------------------------------------
// Generic bf16 WMMA GEMM:  C[M,N] = A[M,K] * B[K,N] + bias[N]
// A,B row-major bf16; C fp32 (ldc); optional bf16 mirror Cb (ldcb).
// Requires K % 32 == 0, N % 4 == 0. 128 threads = 4 waves, 64x64 tile.
// LDS layouts are fragment-native: every WMMA operand = 2x ds_load_b128.
// ---------------------------------------------------------------------------
#define BM 64
#define BN 64
#define BK 32
#define LDA_S 40   // u16 row stride for A tile [m][k]   (80B rows, 16B aligned)
#define LDB_T 40   // u16 row stride for B tile [n][k]   (transposed staging)

__device__ __forceinline__ void stage_tile(const u16* __restrict__ A, int lda,
                                           const u16* __restrict__ Bm, int ldb,
                                           int M, int N, int tileM, int tileN, int k0,
                                           u16* __restrict__ As, u16* __restrict__ Bst,
                                           int tid)
{
    // ---- A tile (64x32 bf16), row-major: 2 chunks of 8 bf16 (b128) per thread
    #pragma unroll
    for (int it = 0; it < 2; ++it) {
        int c = tid + it * 128;
        int m = c >> 2;
        int k = (c & 3) * 8;
        uint4 v = make_uint4(0u, 0u, 0u, 0u);
        int grow = tileM + m;
        if (grow < M) v = *(const uint4*)(A + (size_t)grow * lda + k0 + k);
        *(uint4*)(&As[m * LDA_S + k]) = v;
    }
    // ---- B tile (32K x 64N bf16) -> LDS transposed [n][k].
    //      Each thread: one 2K x 4N sub-block = 2 coalesced b64 loads,
    //      packed into 4 (k,k+1)-pair dwords, 4 ds_store_b32.
    #pragma unroll
    for (int it = 0; it < 2; ++it) {
        int s = tid + it * 128;
        int kb = (s >> 4) * 2;          // 0,2,...,30
        int nb = (s & 15) * 4;          // 0,4,...,60
        uint2 r0 = make_uint2(0u, 0u), r1 = make_uint2(0u, 0u);
        if (tileN + nb + 4 <= N) {
            const u16* p = Bm + (size_t)(k0 + kb) * ldb + tileN + nb;
            r0 = *(const uint2*)p;
            r1 = *(const uint2*)(p + ldb);
        }
        unsigned int p0 = (r0.x & 0xFFFFu) | (r1.x << 16);
        unsigned int p1 = (r0.x >> 16)     | (r1.x & 0xFFFF0000u);
        unsigned int p2 = (r0.y & 0xFFFFu) | (r1.y << 16);
        unsigned int p3 = (r0.y >> 16)     | (r1.y & 0xFFFF0000u);
        *(unsigned int*)(&Bst[(nb + 0) * LDB_T + kb]) = p0;
        *(unsigned int*)(&Bst[(nb + 1) * LDB_T + kb]) = p1;
        *(unsigned int*)(&Bst[(nb + 2) * LDB_T + kb]) = p2;
        *(unsigned int*)(&Bst[(nb + 3) * LDB_T + kb]) = p3;
    }
}

__device__ __forceinline__ void mma_tile(const u16* __restrict__ As,
                                         const u16* __restrict__ Bst,
                                         int warpM, int warpN, int half16, int l16,
                                         v8f acc[2][2])
{
    Frag a[2], b[2];
    // A fragment: lane(half h, m) needs K = [8h,8h+8) U [16+8h,16+8h+8)
    #pragma unroll
    for (int i = 0; i < 2; ++i) {
        const u16* row = As + (warpM + i * 16 + l16) * LDA_S + 8 * half16;
        *(uint4*)(&a[i].d[0]) = *(const uint4*)(row);
        *(uint4*)(&a[i].d[4]) = *(const uint4*)(row + 16);
    }
    // B fragment: lane(half h, n) needs K = [8h,8h+8) U [16+8h,16+8h+8) in [n][k]
    #pragma unroll
    for (int j = 0; j < 2; ++j) {
        const u16* col = Bst + (warpN + j * 16 + l16) * LDB_T + 8 * half16;
        *(uint4*)(&b[j].d[0]) = *(const uint4*)(col);
        *(uint4*)(&b[j].d[4]) = *(const uint4*)(col + 16);
    }
    #pragma unroll
    for (int i = 0; i < 2; ++i)
        #pragma unroll
        for (int j = 0; j < 2; ++j)
            acc[i][j] = __builtin_amdgcn_wmma_f32_16x16x32_bf16(
                false, a[i].v, false, b[j].v, (short)0, acc[i][j], false, false);
}

__global__ __launch_bounds__(128)
void gemm_bf16_wmma(const u16* __restrict__ A, int lda,
                    const u16* __restrict__ Bm, int ldb,
                    const float* __restrict__ bias,
                    float* __restrict__ C, int ldc,
                    u16* __restrict__ Cb, int ldcb,
                    int M, int N, int K)
{
    __shared__ __align__(16) u16 As[2][BM * LDA_S];
    __shared__ __align__(16) u16 Bst[2][BN * LDB_T];

    const int tid   = threadIdx.x;
    const int wave  = tid >> 5;
    const int lane  = tid & 31;
    const int warpM = (wave >> 1) * 32;   // 0 or 32
    const int warpN = (wave & 1) * 32;    // 0 or 32
    const int tileM = blockIdx.y * BM;
    const int tileN = blockIdx.x * BN;
    const int half16 = lane >> 4;
    const int l16    = lane & 15;

    v8f acc[2][2];
    #pragma unroll
    for (int i = 0; i < 2; ++i)
        #pragma unroll
        for (int j = 0; j < 2; ++j)
            #pragma unroll
            for (int r = 0; r < 8; ++r) acc[i][j][r] = 0.0f;

    const int kTiles = K / BK;

    // prologue: stage tile 0
    stage_tile(A, lda, Bm, ldb, M, N, tileM, tileN, 0, As[0], Bst[0], tid);
    __syncthreads();

    for (int kt = 0; kt < kTiles; ++kt) {
        if (kt + 2 < kTiles) {  // hint tile kt+2 into cache (global_prefetch_b8)
            __builtin_prefetch(A + (size_t)tileM * lda + (kt + 2) * BK, 0, 1);
            __builtin_prefetch(Bm + (size_t)((kt + 2) * BK) * ldb + tileN, 0, 1);
        }
        if (kt + 1 < kTiles)    // overlap next tile's loads with this tile's WMMAs
            stage_tile(A, lda, Bm, ldb, M, N, tileM, tileN, (kt + 1) * BK,
                       As[(kt + 1) & 1], Bst[(kt + 1) & 1], tid);

        mma_tile(As[kt & 1], Bst[kt & 1], warpM, warpN, half16, l16, acc);
        __syncthreads();
    }

    // ---- epilogue: bias + store (C layout: VGPR r -> row r + 8*half, col l16)
    #pragma unroll
    for (int i = 0; i < 2; ++i) {
        #pragma unroll
        for (int j = 0; j < 2; ++j) {
            #pragma unroll
            for (int r = 0; r < 8; ++r) {
                int row = tileM + warpM + i * 16 + r + 8 * half16;
                int col = tileN + warpN + j * 16 + l16;
                if (row < M && col < N) {
                    float v = acc[i][j][r] + bias[col];
                    C[(size_t)row * ldc + col] = v;
                    if (Cb) Cb[(size_t)row * ldcb + col] = f2bf(v);
                }
            }
        }
    }
}

// ---------------------------------------------------------------------------
// f32 -> bf16 convert
// ---------------------------------------------------------------------------
__global__ void cvt_kernel(const float* __restrict__ src, u16* __restrict__ dst, long n) {
    long i = (long)blockIdx.x * blockDim.x + threadIdx.x;
    if (i < n) dst[i] = f2bf(src[i]);
}

// ---------------------------------------------------------------------------
// Batched transpose + convert: src[b][R][C] f32 -> dst[b][C][R] bf16
// ---------------------------------------------------------------------------
__global__ __launch_bounds__(256)
void transpose_cvt_kernel(const float* __restrict__ src, u16* __restrict__ dst, int R, int C) {
    __shared__ float tile[32][33];
    const int bz = blockIdx.z;
    src += (size_t)bz * R * C;
    dst += (size_t)bz * R * C;
    const int c0 = blockIdx.x * 32, r0 = blockIdx.y * 32;
    const int tx = threadIdx.x & 31, ty = threadIdx.x >> 5;
    for (int j = ty; j < 32; j += 8) {
        int r = r0 + j, c = c0 + tx;
        if (r < R && c < C) tile[j][tx] = src[(size_t)r * C + c];
    }
    __syncthreads();
    for (int j = ty; j < 32; j += 8) {
        int c = c0 + j, r = r0 + tx;
        if (c < C && r < R) dst[(size_t)c * R + r] = f2bf(tile[tx][j]);
    }
}

// ---------------------------------------------------------------------------
// mean over P of cnn[B,F,P] -> bf16 [B,F]
// ---------------------------------------------------------------------------
__global__ void mean_cvt_kernel(const float* __restrict__ cnn, u16* __restrict__ out) {
    int idx = blockIdx.x * blockDim.x + threadIdx.x;
    if (idx >= kB * kF) return;
    const float* p = cnn + (size_t)idx * kP;
    float s = 0.f;
    for (int i = 0; i < kP; ++i) s += p[i];
    out[idx] = f2bf(s * (1.0f / (float)kP));
}

// ---------------------------------------------------------------------------
// Fused Bahdanau attention per batch row:
//   e = leaky(att1 + att2) @ w_full + b_full ; softmax ; attn = alpha @ img
// One workgroup (256 threads) per b. Writes alpha into d_out's alphas slice.
// ---------------------------------------------------------------------------
__global__ __launch_bounds__(256)
void attention_kernel(const float* __restrict__ att1,   // [B*P, A] f32
                      const float* __restrict__ att2,   // [B, A] f32
                      const float* __restrict__ w_full, // [A]
                      const float* __restrict__ b_full, // [1]
                      const u16*  __restrict__ imgT,    // [B*P, F] bf16
                      u16*        __restrict__ attnb,   // [B, F] bf16
                      float*      __restrict__ alphas,  // [B, T, P]
                      int t)
{
    __shared__ float s_a2[kA];
    __shared__ float s_w[kA];
    __shared__ float s_alpha[kP];
    __shared__ float s_red[256];
    const int b = blockIdx.x, tid = threadIdx.x;

    for (int j = tid; j < kA; j += 256) { s_a2[j] = att2[b * kA + j]; s_w[j] = w_full[j]; }
    __syncthreads();

    float e = 0.f;
    if (tid < kP) {
        const float* row = att1 + ((size_t)b * kP + tid) * kA;
        float acc = b_full[0];
        for (int j = 0; j < kA; ++j) {
            float x = row[j] + s_a2[j];
            acc += (x >= 0.f ? x : 0.2f * x) * s_w[j];
        }
        e = acc;
    }
    s_red[tid] = (tid < kP) ? e : -3.0e38f;
    __syncthreads();
    for (int s = 128; s > 0; s >>= 1) {
        if (tid < s) s_red[tid] = fmaxf(s_red[tid], s_red[tid + s]);
        __syncthreads();
    }
    const float mx = s_red[0];
    __syncthreads();
    float ex = (tid < kP) ? __expf(e - mx) : 0.f;
    s_red[tid] = ex;
    __syncthreads();
    for (int s = 128; s > 0; s >>= 1) {
        if (tid < s) s_red[tid] += s_red[tid + s];
        __syncthreads();
    }
    const float inv = 1.f / s_red[0];
    if (tid < kP) {
        float al = ex * inv;
        s_alpha[tid] = al;
        alphas[((size_t)b * kT + t) * kP + tid] = al;
    }
    __syncthreads();

    // attn_feat[f] = sum_p alpha[p] * img[b,p,f]   (img bf16, L2-resident)
    for (int f = tid; f < kF; f += 256) {
        float acc = 0.f;
        const u16* base = imgT + (size_t)b * kP * kF + f;
        for (int p = 0; p < kP; ++p) acc += s_alpha[p] * bf2f(base[(size_t)p * kF]);
        attnb[b * kF + f] = f2bf(acc);
    }
}

// ---------------------------------------------------------------------------
// Caption embedding gather for timestep t -> first half of x_bf16 [B, 2E]
// ---------------------------------------------------------------------------
__global__ void gather_caption_kernel(const int* __restrict__ cap,
                                      const float* __restrict__ emb,
                                      u16* __restrict__ xb, int t)
{
    const int b = blockIdx.x, tid = threadIdx.x;
    const int tok = cap[b * kT + t];
    const float* row = emb + (size_t)tok * kE;
    for (int j = tid; j < kE; j += 256) xb[b * 2 * kE + j] = f2bf(row[j]);
}

// ---------------------------------------------------------------------------
// GRU elementwise update (PyTorch r,z,n gate order); writes h f32 + bf16
// ---------------------------------------------------------------------------
__global__ void gru_kernel(const float* __restrict__ gi, const float* __restrict__ gh,
                           float* __restrict__ h, u16* __restrict__ hb)
{
    int idx = blockIdx.x * blockDim.x + threadIdx.x;
    if (idx >= kB * kH) return;
    const int b = idx >> 9, j = idx & (kH - 1);
    const float* gib = gi + (size_t)b * 3 * kH;
    const float* ghb = gh + (size_t)b * 3 * kH;
    float r = 1.f / (1.f + __expf(-(gib[j] + ghb[j])));
    float z = 1.f / (1.f + __expf(-(gib[kH + j] + ghb[kH + j])));
    float n = tanhf(gib[2 * kH + j] + r * ghb[2 * kH + j]);
    float hv = h[idx];
    float hn = (1.f - z) * n + z * hv;
    h[idx] = hn;
    hb[idx] = f2bf(hn);
}

// ---------------------------------------------------------------------------
// Host orchestration
// ---------------------------------------------------------------------------
extern "C" void kernel_launch(void* const* d_in, const int* in_sizes, int n_in,
                              void* d_out, int out_size, void* d_ws, size_t ws_size,
                              hipStream_t stream) {
    const float* cnn      = (const float*)d_in[0];
    const int*   cap      = (const int*)  d_in[1];
    const float* emb      = (const float*)d_in[3];
    const float* W_enc    = (const float*)d_in[4];
    const float* b_enc    = (const float*)d_in[5];
    const float* W_dec    = (const float*)d_in[6];
    const float* b_dec    = (const float*)d_in[7];
    const float* w_full   = (const float*)d_in[8];
    const float* b_full   = (const float*)d_in[9];
    const float* W_init   = (const float*)d_in[10];
    const float* b_init   = (const float*)d_in[11];
    const float* W_embed  = (const float*)d_in[12];
    const float* b_embed  = (const float*)d_in[13];
    const float* W_ih     = (const float*)d_in[14];
    const float* b_ih     = (const float*)d_in[15];
    const float* W_hh     = (const float*)d_in[16];
    const float* b_hh     = (const float*)d_in[17];
    const float* W_out    = (const float*)d_in[18];
    const float* b_out    = (const float*)d_in[19];

    float* packed = (float*)d_out;                        // [T*B, V]
    float* alphas = packed + (size_t)kT * kB * kV;        // [B, T, P]

    // ---- carve workspace (256B aligned) ----
    char* ws = (char*)d_ws;
    size_t off = 0;
    auto carve = [&](size_t bytes) -> void* {
        size_t o = (off + 255) & ~(size_t)255;
        off = o + bytes;
        return (void*)(ws + o);
    };
    u16*   imgT   = (u16*)  carve(sizeof(u16) * (size_t)kB * kP * kF);
    u16*   Wencb  = (u16*)  carve(sizeof(u16) * (size_t)kF * kA);
    u16*   Winitb = (u16*)  carve(sizeof(u16) * (size_t)kF * kH);
    u16*   Wembb  = (u16*)  carve(sizeof(u16) * (size_t)kF * kE);
    u16*   Wdecb  = (u16*)  carve(sizeof(u16) * (size_t)kH * kA);
    u16*   WihT   = (u16*)  carve(sizeof(u16) * (size_t)2 * kE * 3 * kH);
    u16*   WhhT   = (u16*)  carve(sizeof(u16) * (size_t)kH * 3 * kH);
    u16*   Woutb  = (u16*)  carve(sizeof(u16) * (size_t)kH * kV);
    u16*   meanb  = (u16*)  carve(sizeof(u16) * (size_t)kB * kF);
    float* att1   = (float*)carve(sizeof(float) * (size_t)kB * kP * kA);
    float* att2   = (float*)carve(sizeof(float) * (size_t)kB * kA);
    float* h      = (float*)carve(sizeof(float) * (size_t)kB * kH);
    u16*   hb     = (u16*)  carve(sizeof(u16) * (size_t)kB * kH);
    u16*   attnb  = (u16*)  carve(sizeof(u16) * (size_t)kB * kF);
    float* emb32  = (float*)carve(sizeof(float) * (size_t)kB * kE);
    u16*   xb     = (u16*)  carve(sizeof(u16) * (size_t)kB * 2 * kE);
    float* gi     = (float*)carve(sizeof(float) * (size_t)kB * 3 * kH);
    float* gh     = (float*)carve(sizeof(float) * (size_t)kB * 3 * kH);
    (void)ws_size; (void)in_sizes; (void)n_in; (void)out_size;

    const dim3 blk256(256), blk128(128);
    auto cvt = [&](const float* s, u16* d, long n) {
        cvt_kernel<<<dim3((unsigned)((n + 255) / 256)), blk256, 0, stream>>>(s, d, n);
    };

    // ---- one-time operand prep (bf16) ----
    cvt(W_enc,   Wencb,  (long)kF * kA);
    cvt(W_init,  Winitb, (long)kF * kH);
    cvt(W_embed, Wembb,  (long)kF * kE);
    cvt(W_dec,   Wdecb,  (long)kH * kA);
    cvt(W_out,   Woutb,  (long)kH * kV);
    transpose_cvt_kernel<<<dim3((2 * kE + 31) / 32, (3 * kH + 31) / 32, 1), blk256, 0, stream>>>(
        W_ih, WihT, 3 * kH, 2 * kE);                              // [3H,2E] -> [2E,3H]
    transpose_cvt_kernel<<<dim3((kH + 31) / 32, (3 * kH + 31) / 32, 1), blk256, 0, stream>>>(
        W_hh, WhhT, 3 * kH, kH);                                  // [3H,H] -> [H,3H]
    transpose_cvt_kernel<<<dim3((kP + 31) / 32, kF / 32, kB), blk256, 0, stream>>>(
        cnn, imgT, kF, kP);                                       // [B,F,P] -> [B,P,F]
    mean_cvt_kernel<<<dim3((kB * kF) / 256), blk256, 0, stream>>>(cnn, meanb);

    // ---- att1 = imgT @ W_enc + b_enc  (12544 x 512 x 2048) ----
    gemm_bf16_wmma<<<dim3(kA / 64, (kB * kP) / 64), blk128, 0, stream>>>(
        imgT, kF, Wencb, kA, b_enc, att1, kA, nullptr, 0, kB * kP, kA, kF);
    // ---- h0 = mean @ W_init + b_init  (64 x 512 x 2048) ----
    gemm_bf16_wmma<<<dim3(kH / 64, 1), blk128, 0, stream>>>(
        meanb, kF, Winitb, kH, b_init, h, kH, hb, kH, kB, kH, kF);

    // ---- recurrent steps ----
    for (int t = 0; t < kT; ++t) {
        // att2 = h @ W_dec + b_dec (64x512x512)
        gemm_bf16_wmma<<<dim3(kA / 64, 1), blk128, 0, stream>>>(
            hb, kH, Wdecb, kA, b_dec, att2, kA, nullptr, 0, kB, kA, kH);
        // attention + softmax + weighted feature sum; alpha -> d_out
        attention_kernel<<<dim3(kB), blk256, 0, stream>>>(
            att1, att2, w_full, b_full, imgT, attnb, alphas, t);
        // x[:, :E] = emb_table[caption[:, t]]
        gather_caption_kernel<<<dim3(kB), blk256, 0, stream>>>(cap, emb, xb, t);
        // x[:, E:] = attn @ W_embed + b_embed  (64x512x2048), bf16 mirror into xb
        gemm_bf16_wmma<<<dim3(kE / 64, 1), blk128, 0, stream>>>(
            attnb, kF, Wembb, kE, b_embed, emb32, kE, xb + kE, 2 * kE, kB, kE, kF);
        // gi = x @ W_ih^T + b_ih  (64x1536x1024)
        gemm_bf16_wmma<<<dim3((3 * kH) / 64, 1), blk128, 0, stream>>>(
            xb, 2 * kE, WihT, 3 * kH, b_ih, gi, 3 * kH, nullptr, 0, kB, 3 * kH, 2 * kE);
        // gh = h @ W_hh^T + b_hh  (64x1536x512)
        gemm_bf16_wmma<<<dim3((3 * kH) / 64, 1), blk128, 0, stream>>>(
            hb, kH, WhhT, 3 * kH, b_hh, gh, 3 * kH, nullptr, 0, kB, 3 * kH, kH);
        // h = GRU(gi, gh, h)
        gru_kernel<<<dim3((kB * kH) / 256), blk256, 0, stream>>>(gi, gh, h, hb);
        // packed[t*B:(t+1)*B, :] = h @ W_out + b_out  (64x30000x512)
        gemm_bf16_wmma<<<dim3((kV + 63) / 64, 1), blk128, 0, stream>>>(
            hb, kH, Woutb, kV, b_out, packed + (size_t)t * kB * kV, kV,
            nullptr, 0, kB, kV, kH);
    }
}
